// IRNN_46127948759577
// MI455X (gfx1250) — matrix operand order repaired and checked by reading, not compiled, not run on hardware
//
#include <hip/hip_runtime.h>
#include <stdint.h>

#define NF      51
#define TT      64
#define TENC    50      // T - LAG
#define UNITS   128
#define G3      384
#define OUTC    102
#define NPAD    112
#define GAMMA   28
#define BT      64      // batch rows per block
#define NTHREADS 256
#define KW      13056   // UNITS*OUTC
#define SROW    52      // staging row stride (floats)
#define OP_SCALE 0.05f
#define Q_SCALE  0.02f
#define CC       0.5413248546129181f  // log(expm1(1))

typedef __attribute__((ext_vector_type(16))) __bf16 v16bf;
typedef __attribute__((ext_vector_type(8)))  __bf16 v8bf;
typedef __attribute__((ext_vector_type(8)))  float  v8f;

__device__ __forceinline__ float softplusf(float v){
  return (v > 20.0f) ? v : log1pf(__expf(v));
}
__device__ __forceinline__ float sigmoidf(float v){
  return 1.0f / (1.0f + __expf(-v));
}
__device__ __forceinline__ uint32_t hmix(uint32_t x){
  x ^= x >> 16; x *= 0x7feb352du; x ^= x >> 15; x *= 0x846ca68bu; x ^= x >> 16; return x;
}
// deterministic standard-normal surrogate (Box-Muller on hashed uniforms)
__device__ __forceinline__ float rnormf(uint32_t a, uint32_t b){
  uint32_t h1 = hmix(a * 0x9E3779B9u + b * 0x85EBCA6Bu + 0x165667B1u);
  uint32_t h2 = hmix(h1 ^ 0x27220A95u);
  float u1 = (float)(h1 >> 8) * (1.0f/16777216.0f) + 1.0e-7f;
  float u2 = (float)(h2 >> 8) * (1.0f/16777216.0f);
  return sqrtf(-2.0f * __logf(u1)) * __cosf(6.28318530718f * u2);
}

__device__ __forceinline__ v8f wmma_bf16(v16bf a, v16bf b, v8f c){
  return __builtin_amdgcn_wmma_f32_16x16x32_bf16(false, a, false, b, (short)0, c,
                                                 false, false);
}

// A-operand (16x32 bf16) fragment from row-major LDS tile.
// Lanes 0-15: M=lane,   K = k0..k0+7  and  k0+16..k0+23
// Lanes16-31: M=lane-16,K = k0+8..k0+15 and k0+24..k0+31  (k0 = ks*32)
__device__ __forceinline__ v16bf load_a(const __bf16* base, int stride, int mt,
                                        int ks, int lane){
  int m  = mt*16 + (lane & 15);
  int k0 = ks*32 + ((lane >> 4) << 3);
  const __bf16* p = base + m*stride + k0;
  v8bf lo = *(const v8bf*)(p);
  v8bf hi = *(const v8bf*)(p + 16);
  v16bf a;
  #pragma unroll
  for (int i = 0; i < 8; ++i){ a[i] = lo[i]; a[i+8] = hi[i]; }
  return a;
}

// B-operand (32x16 bf16) fragment pre-swizzled in LDS: 512 halfs per frag,
// lane-major (lane l holds its 16 contiguous halfs -> 2x ds_load_b128).
__device__ __forceinline__ v16bf load_b(const __bf16* fr, int fidx, int lane){
  return *(const v16bf*)(fr + ((fidx * 32 + lane) << 4));
}

__global__ __launch_bounds__(NTHREADS, 1)
void irnn_kernel(const float* __restrict__ x,
                 const float* __restrict__ gk,     // 51 x 384
                 const float* __restrict__ gr,     // 128 x 384
                 const float* __restrict__ gb,     // 2 x 384
                 const float* __restrict__ ploc,   // 13158
                 const float* __restrict__ prho,   // 13158
                 float* __restrict__ out)          // 8192 x 28 x 102
{
  __shared__ __bf16 sWk[2*24*512];     // input kernel, K padded 51->64
  __shared__ __bf16 sWr[4*24*512];     // recurrent kernel, K=128
  __shared__ __bf16 sWo[4*7*512];      // sampled readout kernel, N padded 102->112
  __shared__ __bf16 sX[BT*64];         // current input tile (padded to 64 cols)
  __shared__ __bf16 sH[BT*UNITS];      // hidden state (A-operand for recurrent GEMM)
  __shared__ float  sT[BT*NPAD];       // readout result t
  __shared__ float  sStage[2][BT*SROW];// async double-buffer for x tiles (f32)
  __shared__ float  sBz[UNITS], sBr[UNITS], sBhx[UNITS], sBhh[UNITS];
  __shared__ float  sBo[NPAD];

  const int tid  = threadIdx.x;
  const int w    = tid >> 5;      // wave id 0..7
  const int lane = tid & 31;
  const int rowbase = blockIdx.x * BT;

  // ---- pack gru_kernel into B-fragment layout (zero-pad K 51->64) ----
  for (int idx = tid; idx < 2*24*512; idx += NTHREADS){
    int f = idx >> 9, wi = idx & 511, l = wi >> 4, i = wi & 15;
    int kt = f / 24, nt = f - kt*24;
    int k = kt*32 + ((l >> 4) << 3) + (i & 7) + ((i >> 3) << 4);
    int n = nt*16 + (l & 15);
    sWk[idx] = (__bf16)((k < NF) ? gk[k*G3 + n] : 0.0f);
  }
  // ---- pack gru_rkernel ----
  for (int idx = tid; idx < 4*24*512; idx += NTHREADS){
    int f = idx >> 9, wi = idx & 511, l = wi >> 4, i = wi & 15;
    int kt = f / 24, nt = f - kt*24;
    int k = kt*32 + ((l >> 4) << 3) + (i & 7) + ((i >> 3) << 4);
    int n = nt*16 + (l & 15);
    sWr[idx] = (__bf16)gr[k*G3 + n];
  }
  // ---- gate biases (z,r fold both biases; h-gate keeps them split) ----
  for (int u = tid; u < UNITS; u += NTHREADS){
    sBz[u]  = gb[u]       + gb[G3 + u];
    sBr[u]  = gb[128 + u] + gb[G3 + 128 + u];
    sBhx[u] = gb[256 + u];
    sBhh[u] = gb[G3 + 256 + u];
  }
  for (int idx = tid; idx < BT*64;    idx += NTHREADS) sX[idx] = (__bf16)0.0f;
  for (int idx = tid; idx < BT*UNITS; idx += NTHREADS) sH[idx] = (__bf16)0.0f;

  v8f hreg[4];
  #pragma unroll
  for (int mt = 0; mt < 4; ++mt)
    #pragma unroll
    for (int i = 0; i < 8; ++i) hreg[mt][i] = 0.0f;

  __syncthreads();

  const int ucol = w*16 + (lane & 15);   // hidden unit owned by this lane
  const float bz = sBz[ucol], br = sBr[ucol], bhx = sBhx[ucol], bhh = sBhh[ucol];

  // ---- async-copy one timestep's x tile (f32) into an LDS staging buffer ----
  // GLOBAL_LOAD_ASYNC_TO_LDS_B32, GVS mode: mem = SGPR64(base) + VGPR32(off)
  auto issue_async = [&](int t, int buf){
    for (int idx = tid; idx < BT*NF; idx += NTHREADS){
      int row = idx / NF, f = idx - row*NF;
      uint32_t ldsaddr = (uint32_t)(uintptr_t)&sStage[buf][row*SROW + f];
      uint32_t goff = (uint32_t)((((rowbase + row)*(TT*NF)) + t*NF + f) * 4);
      asm volatile("global_load_async_to_lds_b32 %0, %1, %2"
                   :: "v"(ldsaddr), "v"(goff), "s"(x) : "memory");
    }
  };
  auto wait_async = [&](){
    asm volatile("s_wait_asynccnt 0" ::: "memory");
  };

  // ---------------- fused GRU step (wave w owns units [w*16, w*16+16)) -----
  auto gru_step = [&](){
    v8f az[4], ar[4], ahx[4], ahh[4];
    #pragma unroll
    for (int mt = 0; mt < 4; ++mt)
      #pragma unroll
      for (int i = 0; i < 8; ++i){ az[mt][i]=bz; ar[mt][i]=br; ahx[mt][i]=bhx; ahh[mt][i]=bhh; }
    #pragma unroll
    for (int mt = 0; mt < 4; ++mt){
      #pragma unroll
      for (int ks = 0; ks < 2; ++ks){           // mx = x @ Wk  (K padded to 64)
        v16bf a = load_a(sX, 64, mt, ks, lane);
        az[mt]  = wmma_bf16(a, load_b(sWk, ks*24 + w,      lane), az[mt]);
        ar[mt]  = wmma_bf16(a, load_b(sWk, ks*24 + 8 + w,  lane), ar[mt]);
        ahx[mt] = wmma_bf16(a, load_b(sWk, ks*24 + 16 + w, lane), ahx[mt]);
      }
      #pragma unroll
      for (int ks = 0; ks < 4; ++ks){           // mh = h @ Wr  (K = 128)
        v16bf a = load_a(sH, UNITS, mt, ks, lane);
        az[mt]  = wmma_bf16(a, load_b(sWr, ks*24 + w,      lane), az[mt]);
        ar[mt]  = wmma_bf16(a, load_b(sWr, ks*24 + 8 + w,  lane), ar[mt]);
        ahh[mt] = wmma_bf16(a, load_b(sWr, ks*24 + 16 + w, lane), ahh[mt]);
      }
    }
    __syncthreads();   // all reads of sH done before we overwrite it
    #pragma unroll
    for (int mt = 0; mt < 4; ++mt){
      #pragma unroll
      for (int i = 0; i < 8; ++i){
        float z  = sigmoidf(az[mt][i]);
        float r  = sigmoidf(ar[mt][i]);
        float hh = tanhf(ahx[mt][i] + r * ahh[mt][i]);
        float hn = z * hreg[mt][i] + (1.0f - z) * hh;
        hreg[mt][i] = hn;
        int row = mt*16 + i + ((lane >> 4) << 3);   // C-layout row
        sH[row*UNITS + ucol] = (__bf16)hn;
      }
    }
    __syncthreads();
  };

  // ---- build sampled readout weights for sample-step g ----
  auto build_wo = [&](int g){
    for (int idx = tid; idx < 4*7*512; idx += NTHREADS){
      int f = idx >> 9, wi2 = idx & 511, l = wi2 >> 4, i = wi2 & 15;
      int kt = f / 7, nt = f - kt*7;
      int k = kt*32 + ((l >> 4) << 3) + (i & 7) + ((i >> 3) << 4);
      int n = nt*16 + (l & 15);
      float v = 0.0f;
      if (n < OUTC){
        int widx = k*OUTC + n;
        float sc = 1e-5f + Q_SCALE * softplusf(CC + prho[widx]);
        v = ploc[widx] + sc * rnormf(0xA11CEu + (uint32_t)g, (uint32_t)widx);
      }
      sWo[idx] = (__bf16)v;
    }
    for (int n = tid; n < NPAD; n += NTHREADS){
      float v = 0.0f;
      if (n < OUTC){
        int widx = KW + n;
        float sc = 1e-5f + Q_SCALE * softplusf(CC + prho[widx]);
        v = ploc[widx] + sc * rnormf(0xA11CEu + (uint32_t)g, (uint32_t)widx);
      }
      sBo[n] = v;
    }
  };

  // ---- t = h @ Wo + bo   (M=64, N=112, K=128) -> sT ----
  auto dv_gemm = [&](){
    for (int tt = w; tt < 28; tt += 8){
      int mt = tt & 3, nt = tt >> 2;
      float bo = sBo[nt*16 + (lane & 15)];
      v8f acc;
      #pragma unroll
      for (int i = 0; i < 8; ++i) acc[i] = bo;
      #pragma unroll
      for (int ks = 0; ks < 4; ++ks){
        v16bf a = load_a(sH, UNITS, mt, ks, lane);
        acc = wmma_bf16(a, load_b(sWo, ks*7 + nt, lane), acc);
      }
      #pragma unroll
      for (int i = 0; i < 8; ++i){
        int row = mt*16 + i + ((lane >> 4) << 3);
        sT[row*NPAD + nt*16 + (lane & 15)] = acc[i];
      }
    }
  };

  auto write_out = [&](int g){
    for (int idx = tid; idx < BT*OUTC; idx += NTHREADS){
      int row = idx / OUTC, c = idx - row*OUTC;
      float raw = sT[row*NPAD + c];
      float v = (c < NF) ? raw : (1e-5f + OP_SCALE * softplusf(CC + raw));
      out[((size_t)(rowbase + row))*(GAMMA*OUTC) + (size_t)g*OUTC + c] = v;
    }
  };

  // ======================= encoder: 50 GRU steps =========================
  // Software pipeline: async-copy x tile for step t+1 into LDS while the
  // GRU WMMAs for step t run; ASYNCcnt + barrier makes it visible.
  issue_async(0, 0);
  for (int t = 0; t < TENC; ++t){
    int cur = t & 1, nxt = cur ^ 1;
    wait_async();          // wave's own async loads (issued last iteration)
    __syncthreads();       // cross-wave visibility of staged tile
    for (int idx = tid; idx < BT*NF; idx += NTHREADS){
      int row = idx / NF, f = idx - row*NF;
      sX[row*64 + f] = (__bf16)sStage[cur][row*SROW + f];
    }
    if (t + 1 < TENC) issue_async(t + 1, nxt);
    __syncthreads();
    gru_step();
  }

  // ======================= first sample (g = 0) ==========================
  build_wo(0);
  __syncthreads();
  dv_gemm();
  __syncthreads();
  write_out(0);

  // ======================= decoder: 27 steps =============================
  for (int g = 1; g < GAMMA; ++g){
    // s = loc + sc * eps_s  -> becomes next GRU input
    for (int idx = tid; idx < BT*NF; idx += NTHREADS){
      int row = idx / NF, f = idx - row*NF;
      float loc = sT[row*NPAD + f];
      float sc  = 1e-5f + OP_SCALE * softplusf(CC + sT[row*NPAD + NF + f]);
      float es  = rnormf(0x5EED5u + (uint32_t)(g - 1),
                         (uint32_t)((rowbase + row)*NF + f));
      sX[row*64 + f] = (__bf16)(loc + sc * es);
    }
    __syncthreads();
    gru_step();
    build_wo(g);
    __syncthreads();
    dv_gemm();
    __syncthreads();
    write_out(g);
  }
}

extern "C" void kernel_launch(void* const* d_in, const int* in_sizes, int n_in,
                              void* d_out, int out_size, void* d_ws, size_t ws_size,
                              hipStream_t stream) {
  (void)in_sizes; (void)n_in; (void)d_ws; (void)ws_size; (void)out_size;
  const float* x    = (const float*)d_in[0];
  const float* gk   = (const float*)d_in[1];
  const float* gr   = (const float*)d_in[2];
  const float* gb   = (const float*)d_in[3];
  const float* ploc = (const float*)d_in[4];
  const float* prho = (const float*)d_in[5];
  float* out = (float*)d_out;
  dim3 grid(8192 / BT);   // 128 workgroups, one per WGP, full sequence each
  irnn_kernel<<<grid, NTHREADS, 0, stream>>>(x, gk, gr, gb, ploc, prho, out);
}
